// EnhancedDeepGTAT_GAT_65884798320879
// MI455X (gfx1250) — matrix operand order repaired and checked by reading, not compiled, not run on hardware
//
#include <hip/hip_runtime.h>
#include <hip/hip_bf16.h>

// ---------------------------------------------------------------------------
// EnhancedDeepGTAT (4x GAT + LSTM scan + linear) for MI455X / gfx1250.
// - All GEMMs: bf16 WMMA (v_wmma_f32_16x16x32_bf16), f32 accumulate,
//   software-pipelined (register double-buffered fragments, k unrolled by 64)
//   so global_load_b128 for k+32 overlaps the WMMAs for k.
// - GAT softmax/aggregate: CSR (built once) gather, fused with LayerNorm+ELU,
//   with global_prefetch of the next source row.
// - LSTM: 512 sequential steps, each one fused [x_t|h] @ [Wih|Whh]^T WMMA GEMM
//   (K=1024) + elementwise gate update. No big precomputed-gates buffer.
// ---------------------------------------------------------------------------

#define N_NODES   16384
#define N_GRAPHS  32
#define SEQ       512          // nodes per graph
#define DEG       16
#define E_EDGES   (N_NODES * DEG)
#define IN_DIM    256
#define HID       512
#define GATES     (4 * HID)    // 2048
#define KCAT      (2 * HID)    // 1024
#define NCLS      10

typedef __attribute__((ext_vector_type(16))) __bf16 v16bf;
typedef __attribute__((ext_vector_type(8)))  float  v8f;

// ---------------- small utility kernels ----------------

__global__ void fill_i32(int* p, int v, int n) {
    int i = blockIdx.x * 256 + threadIdx.x;
    if (i < n) p[i] = v;
}
__global__ void fill_f32(float* p, float v, int n) {
    int i = blockIdx.x * 256 + threadIdx.x;
    if (i < n) p[i] = v;
}
__global__ void cvt_f32_bf16(const float* __restrict__ s, __bf16* __restrict__ d, int n) {
    int i = blockIdx.x * 256 + threadIdx.x;
    if (i < n) d[i] = (__bf16)s[i];
}
// Wcat[n, 0:512] = Wih[n], Wcat[n, 512:1024] = Whh[n]   (bf16)
__global__ void build_wcat(const float* __restrict__ wih, const float* __restrict__ whh,
                           __bf16* __restrict__ wcat) {
    int i = blockIdx.x * 256 + threadIdx.x;          // GATES*KCAT threads
    int n = i >> 10, k = i & 1023;
    float v = (k < HID) ? wih[n * HID + k] : whh[n * HID + (k - HID)];
    wcat[i] = (__bf16)v;
}

__device__ __forceinline__ float wave_sum(float v) {
    #pragma unroll
    for (int off = 16; off > 0; off >>= 1) v += __shfl_xor(v, off);
    return v;
}
__device__ __forceinline__ float wave_max(float v) {
    #pragma unroll
    for (int off = 16; off > 0; off >>= 1) v = fmaxf(v, __shfl_xor(v, off));
    return v;
}
__device__ __forceinline__ float sigmoidf(float x) { return 1.0f / (1.0f + __expf(-x)); }

// ---------------- WMMA fragment helpers ----------------
// A-frag (16x32 bf16, wave32): lane L: m=L&15, half=L>>4,
//   elems 0..15 = A[m][k0 + half*16 + j]
// B-frag (32x16): lane L: n=L&15, elems 0..15 = W[n][k0 + half*16 + j] (contig 32B)

struct Frag {
    union { uint4 q[2]; v16bf v; };
};
__device__ __forceinline__ void load_a(Frag& f, const __bf16* __restrict__ arow,
                                       int k0, int half) {
    f.q[0] = *(const uint4*)(arow + k0 + half * 16);
    f.q[1] = *(const uint4*)(arow + k0 + half * 16 + 8);
}
__device__ __forceinline__ void load_b(Frag& f, const __bf16* __restrict__ wrow,
                                       int k0, int half) {
    const uint4* bp = (const uint4*)(wrow + k0 + half * 16);
    f.q[0] = bp[0];
    f.q[1] = bp[1];
}

// ---------------- WMMA GEMM: C[M,N] = A[M,K](bf16) @ W[N,K]^T(bf16) (+bias) ---
// One wave computes a 16(M) x 64(N) tile: 4 accumulators, A fragment reused.
// C/D layout: lane L: n = L&15; element r <-> m = half*8 + r.
// grid = (M/16, N/(64*8)), block = 256 (8 waves). Requires K % 64 == 0.
// Double-buffered: loads for the next 32-K slab are issued before the WMMAs
// consuming the current slab.

__global__ __launch_bounds__(256) void wmma_gemm_bt(
    const __bf16* __restrict__ A, const __bf16* __restrict__ W,
    const float* __restrict__ bias, float* __restrict__ C,
    int M, int N, int K)
{
    const int wave = threadIdx.x >> 5, lane = threadIdx.x & 31;
    const int tileM = blockIdx.x;
    const int tileN = blockIdx.y * 8 + wave;
    const int ml = lane & 15, half = lane >> 4;
    const __bf16* arow = A + (size_t)(tileM * 16 + ml) * K;
    const __bf16* wrow[4];
    #pragma unroll
    for (int nt = 0; nt < 4; ++nt)
        wrow[nt] = W + (size_t)(tileN * 64 + nt * 16 + ml) * K;

    v8f acc[4] = {};
    Frag a0, a1, b0[4], b1[4];
    load_a(a0, arow, 0, half);
    #pragma unroll
    for (int nt = 0; nt < 4; ++nt) load_b(b0[nt], wrow[nt], 0, half);

    for (int k0 = 0; k0 < K; k0 += 64) {
        // issue loads for slab k0+32 while computing slab k0
        load_a(a1, arow, k0 + 32, half);
        #pragma unroll
        for (int nt = 0; nt < 4; ++nt) load_b(b1[nt], wrow[nt], k0 + 32, half);
        #pragma unroll
        for (int nt = 0; nt < 4; ++nt)
            acc[nt] = __builtin_amdgcn_wmma_f32_16x16x32_bf16(
                false, a0.v, false, b0[nt].v, (short)0, acc[nt], false, false);
        // issue loads for slab k0+64 while computing slab k0+32
        if (k0 + 64 < K) {
            load_a(a0, arow, k0 + 64, half);
            #pragma unroll
            for (int nt = 0; nt < 4; ++nt) load_b(b0[nt], wrow[nt], k0 + 64, half);
        }
        #pragma unroll
        for (int nt = 0; nt < 4; ++nt)
            acc[nt] = __builtin_amdgcn_wmma_f32_16x16x32_bf16(
                false, a1.v, false, b1[nt].v, (short)0, acc[nt], false, false);
    }
    #pragma unroll
    for (int nt = 0; nt < 4; ++nt) {
        int gn = tileN * 64 + nt * 16 + ml;
        float bv = bias ? bias[gn] : 0.0f;
        #pragma unroll
        for (int r = 0; r < 8; ++r) {
            int gm = tileM * 16 + half * 8 + r;
            C[(size_t)gm * N + gn] = acc[nt][r] + bv;
        }
    }
}

// ---------------- GAT per-node attention partial scores ----------------
// s_i[n] = sum_c h[n,c]*ai(c), s_j likewise; c = hh*128+cc, attn[HEADS,2C].
__global__ __launch_bounds__(256) void gat_scores(
    const float* __restrict__ hlin, const float* __restrict__ attn,
    float* __restrict__ s_i, float* __restrict__ s_j)
{
    const int wave = threadIdx.x >> 5, lane = threadIdx.x & 31;
    const int n = blockIdx.x * 8 + wave;
    const float* hr = hlin + (size_t)n * HID;
    float si = 0.f, sj = 0.f;
    #pragma unroll
    for (int j = 0; j < 16; ++j) {
        int c = lane * 16 + j;
        int hh = c >> 7, cc = c & 127;
        float hv = hr[c];
        si += hv * attn[hh * 256 + cc];
        sj += hv * attn[hh * 256 + 128 + cc];
    }
    si = wave_sum(si); sj = wave_sum(sj);
    if (lane == 0) { s_i[n] = si; s_j[n] = sj; }
}

// ---------------- per-edge raw attention score (leaky relu 0.2) --------
__global__ void gat_edge_scores(const int* __restrict__ eidx,
                                const float* __restrict__ s_i,
                                const float* __restrict__ s_j,
                                float* __restrict__ esc)
{
    int e = blockIdx.x * 256 + threadIdx.x;
    if (e >= E_EDGES) return;
    int s = eidx[e], t = eidx[E_EDGES + e];
    float sc = s_i[t] + s_j[s];
    esc[e] = (sc > 0.f) ? sc : 0.2f * sc;
}

// ---------------- CSR build (edges are identical for all layers) -------
__global__ void csr_count(const int* __restrict__ eidx, int* __restrict__ deg) {
    int e = blockIdx.x * 256 + threadIdx.x;
    if (e < E_EDGES) atomicAdd(&deg[eidx[E_EDGES + e]], 1);
}
__global__ __launch_bounds__(1024) void csr_scan(const int* __restrict__ deg,
                                                 int* __restrict__ rowstart) {
    __shared__ int part[1024];
    int t = threadIdx.x, base = t * 16;
    int loc[16]; int s = 0;
    #pragma unroll
    for (int j = 0; j < 16; ++j) { loc[j] = s; s += deg[base + j]; }
    part[t] = s;
    __syncthreads();
    for (int off = 1; off < 1024; off <<= 1) {
        int v = (t >= off) ? part[t - off] : 0;
        __syncthreads();
        part[t] += v;
        __syncthreads();
    }
    int pre = (t == 0) ? 0 : part[t - 1];
    #pragma unroll
    for (int j = 0; j < 16; ++j) rowstart[base + j] = pre + loc[j];
    if (t == 1023) rowstart[N_NODES] = part[1023];
}
__global__ void csr_scatter(const int* __restrict__ eidx, const int* __restrict__ rowstart,
                            int* __restrict__ cursor, int* __restrict__ le, int* __restrict__ ls) {
    int e = blockIdx.x * 256 + threadIdx.x;
    if (e >= E_EDGES) return;
    int s = eidx[e], t = eidx[E_EDGES + e];
    int pos = atomicAdd(&cursor[t], 1);
    int at = rowstart[t] + pos;
    le[at] = e; ls[at] = s;
}

// ---------------- fused: segment softmax + aggregate + (out+res) LN + ELU ----
// One wave per target node; lane owns 16 contiguous channels (float4 loads).
// Prefetches the next source row while accumulating the current one.
__global__ __launch_bounds__(256) void gat_aggregate_ln_elu(
    const float* __restrict__ hlin, const float* __restrict__ outres,
    const float* __restrict__ esc, const int* __restrict__ rowstart,
    const int* __restrict__ le, const int* __restrict__ ls,
    const float* __restrict__ gamma, const float* __restrict__ beta,
    __bf16* __restrict__ actout)
{
    const int wave = threadIdx.x >> 5, lane = threadIdx.x & 31;
    const int n = blockIdx.x * 8 + wave;
    const int row0 = rowstart[n];
    const int cnt = rowstart[n + 1] - row0;

    float m = -3.4e38f;
    for (int i = lane; i < cnt; i += 32) m = fmaxf(m, esc[le[row0 + i]]);
    m = wave_max(m);
    if (cnt == 0) m = 0.f;                       // ref: where(isfinite(m), m, 0)

    float z = 0.f;
    for (int i = lane; i < cnt; i += 32) z += __expf(esc[le[row0 + i]] - m);
    z = wave_sum(z);
    const float inv = 1.0f / (z + 1e-16f);

    float acc[16];
    const float4* rr = (const float4*)(outres + (size_t)n * HID) + lane * 4;
    #pragma unroll
    for (int q = 0; q < 4; ++q) {
        float4 v = rr[q];
        acc[q * 4 + 0] = v.x; acc[q * 4 + 1] = v.y; acc[q * 4 + 2] = v.z; acc[q * 4 + 3] = v.w;
    }
    if (cnt > 0) {
        // warm up: prefetch first source row (global_prefetch_b8)
        __builtin_prefetch(hlin + (size_t)ls[row0] * HID + lane * 16, 0, 3);
    }
    for (int i = 0; i < cnt; ++i) {
        int e = le[row0 + i], s = ls[row0 + i];
        if (i + 1 < cnt)
            __builtin_prefetch(hlin + (size_t)ls[row0 + i + 1] * HID + lane * 16, 0, 3);
        float w = __expf(esc[e] - m) * inv;
        const float4* hr = (const float4*)(hlin + (size_t)s * HID) + lane * 4;
        #pragma unroll
        for (int q = 0; q < 4; ++q) {
            float4 v = hr[q];
            acc[q * 4 + 0] += w * v.x; acc[q * 4 + 1] += w * v.y;
            acc[q * 4 + 2] += w * v.z; acc[q * 4 + 3] += w * v.w;
        }
    }
    float sum = 0.f, sq = 0.f;
    #pragma unroll
    for (int j = 0; j < 16; ++j) { sum += acc[j]; sq += acc[j] * acc[j]; }
    sum = wave_sum(sum); sq = wave_sum(sq);
    const float mu = sum * (1.0f / HID);
    const float var = sq * (1.0f / HID) - mu * mu;
    const float rstd = rsqrtf(var + 1e-5f);

    union { __bf16 h[16]; uint4 u[2]; } ob;
    const int cb = lane * 16;
    #pragma unroll
    for (int j = 0; j < 16; ++j) {
        float y = (acc[j] - mu) * rstd * gamma[cb + j] + beta[cb + j];
        y = (y > 0.f) ? y : (__expf(y) - 1.0f);            // ELU(alpha=1)
        ob.h[j] = (__bf16)y;
    }
    uint4* dst = (uint4*)(actout + (size_t)n * HID + cb);
    dst[0] = ob.u[0]; dst[1] = ob.u[1];
}

// ---------------- LSTM step GEMM: gates[32,2048] = [x_t|h] @ Wcat^T + bih+bhh ----
// grid = (2, 4), block = 256. K = 1024 (first 512 from act row g*SEQ+t, rest
// from h). Each 32-wide K slab lies entirely within one half, so the A-row
// pointer is selected per slab. Double-buffered like wmma_gemm_bt.
__device__ __forceinline__ const __bf16* lstm_arow(const __bf16* xrow,
                                                   const __bf16* hrow, int k0) {
    return (k0 < HID) ? (xrow + k0) : (hrow + (k0 - HID));
}

__global__ __launch_bounds__(256) void lstm_step_gemm(
    const __bf16* __restrict__ act, const __bf16* __restrict__ hb,
    const __bf16* __restrict__ wcat,
    const float* __restrict__ bih, const float* __restrict__ bhh,
    float* __restrict__ gates, int t)
{
    const int wave = threadIdx.x >> 5, lane = threadIdx.x & 31;
    const int tileM = blockIdx.x;                 // 0..1
    const int tileN = blockIdx.y * 8 + wave;      // 0..31
    const int ml = lane & 15, half = lane >> 4;
    const int g = tileM * 16 + ml;
    const __bf16* xrow = act + ((size_t)g * SEQ + t) * HID;
    const __bf16* hrow = hb + (size_t)g * HID;
    const __bf16* wrow[4];
    #pragma unroll
    for (int nt = 0; nt < 4; ++nt)
        wrow[nt] = wcat + (size_t)(tileN * 64 + nt * 16 + ml) * KCAT;

    v8f acc[4] = {};
    Frag a0, a1, b0[4], b1[4];
    load_a(a0, lstm_arow(xrow, hrow, 0), 0, half);
    #pragma unroll
    for (int nt = 0; nt < 4; ++nt) load_b(b0[nt], wrow[nt], 0, half);

    for (int k0 = 0; k0 < KCAT; k0 += 64) {
        load_a(a1, lstm_arow(xrow, hrow, k0 + 32), 0, half);
        #pragma unroll
        for (int nt = 0; nt < 4; ++nt) load_b(b1[nt], wrow[nt], k0 + 32, half);
        #pragma unroll
        for (int nt = 0; nt < 4; ++nt)
            acc[nt] = __builtin_amdgcn_wmma_f32_16x16x32_bf16(
                false, a0.v, false, b0[nt].v, (short)0, acc[nt], false, false);
        if (k0 + 64 < KCAT) {
            load_a(a0, lstm_arow(xrow, hrow, k0 + 64), 0, half);
            #pragma unroll
            for (int nt = 0; nt < 4; ++nt) load_b(b0[nt], wrow[nt], k0 + 64, half);
        }
        #pragma unroll
        for (int nt = 0; nt < 4; ++nt)
            acc[nt] = __builtin_amdgcn_wmma_f32_16x16x32_bf16(
                false, a1.v, false, b1[nt].v, (short)0, acc[nt], false, false);
    }
    #pragma unroll
    for (int nt = 0; nt < 4; ++nt) {
        int gn = tileN * 64 + nt * 16 + ml;
        float bv = bih[gn] + bhh[gn];
        #pragma unroll
        for (int r = 0; r < 8; ++r) {
            int gm = tileM * 16 + half * 8 + r;
            gates[(size_t)gm * GATES + gn] = acc[nt][r] + bv;
        }
    }
}

// ---------------- LSTM elementwise update (torch gate order i,f,g,o) ----
__global__ void lstm_update(const float* __restrict__ gates,
                            float* __restrict__ h, float* __restrict__ c,
                            __bf16* __restrict__ hb)
{
    int idx = blockIdx.x * 256 + threadIdx.x;     // 32*512
    int g = idx >> 9, j = idx & 511;
    const float* gr = gates + (size_t)g * GATES;
    float iv = sigmoidf(gr[j]);
    float fv = sigmoidf(gr[HID + j]);
    float gv = tanhf(gr[2 * HID + j]);
    float ov = sigmoidf(gr[3 * HID + j]);
    float cv = fv * c[idx] + iv * gv;
    float hv = ov * tanhf(cv);
    c[idx] = cv; h[idx] = hv; hb[idx] = (__bf16)hv;
}

// ---------------- classifier: out[32,10] = h @ W^T + b ----------------
__global__ __launch_bounds__(1024) void classifier(
    const float* __restrict__ h, const float* __restrict__ w,
    const float* __restrict__ b, float* __restrict__ out)
{
    const int wave = threadIdx.x >> 5, lane = threadIdx.x & 31;
    const int o = blockIdx.x * 32 + wave;         // 10 blocks * 32 waves = 320
    const int g = o / NCLS, k = o % NCLS;
    const float* hr = h + (size_t)g * HID;
    const float* wr = w + (size_t)k * HID;
    float s = 0.f;
    for (int i = lane; i < HID; i += 32) s += hr[i] * wr[i];
    s = wave_sum(s);
    if (lane == 0) out[o] = s + b[k];
}

// ---------------------------------------------------------------------------

extern "C" void kernel_launch(void* const* d_in, const int* in_sizes, int n_in,
                              void* d_out, int out_size, void* d_ws, size_t ws_size,
                              hipStream_t stream) {
    (void)in_sizes; (void)n_in; (void)out_size; (void)ws_size;

    // ---- inputs (setup_inputs() dict insertion order) ----
    const float* x    = (const float*)d_in[0];
    const int*   eidx = (const int*)d_in[1];
    // d_in[2] = batch (unused; graphs are packed & equal-size)
    const float* lyr_lin_w[4], *lyr_attn[4], *lyr_g[4], *lyr_b[4], *lyr_res_w[4], *lyr_res_b[4];
    for (int l = 0; l < 4; ++l) {
        int base = 3 + 6 * l;
        lyr_lin_w[l] = (const float*)d_in[base + 0];
        lyr_attn[l]  = (const float*)d_in[base + 1];
        lyr_g[l]     = (const float*)d_in[base + 2];
        lyr_b[l]     = (const float*)d_in[base + 3];
        lyr_res_w[l] = (const float*)d_in[base + 4];
        lyr_res_b[l] = (const float*)d_in[base + 5];
    }
    const float* lstm_wih = (const float*)d_in[27];
    const float* lstm_whh = (const float*)d_in[28];
    const float* lstm_bih = (const float*)d_in[29];
    const float* lstm_bhh = (const float*)d_in[30];
    const float* fc_w     = (const float*)d_in[31];
    const float* fc_b     = (const float*)d_in[32];
    float* out = (float*)d_out;

    // ---- workspace carve-out (~93 MB total) ----
    char* w = (char*)d_ws;
    size_t off = 0;
    auto take = [&](size_t bytes) -> void* {
        void* p = w + off;
        off = (off + bytes + 255) & ~(size_t)255;
        return p;
    };
    __bf16* wl_lin[4]; __bf16* wl_res[4];
    for (int l = 0; l < 4; ++l) {
        size_t kin = (l == 0) ? IN_DIM : HID;
        wl_lin[l] = (__bf16*)take(HID * kin * 2);
        wl_res[l] = (__bf16*)take(HID * kin * 2);
    }
    __bf16* wcat   = (__bf16*)take((size_t)GATES * KCAT * 2);
    __bf16* actb   = (__bf16*)take((size_t)N_NODES * HID * 2);
    float*  hlin   = (float*)take((size_t)N_NODES * HID * 4);
    float*  outres = (float*)take((size_t)N_NODES * HID * 4);
    float*  s_i    = (float*)take(N_NODES * 4);
    float*  s_j    = (float*)take(N_NODES * 4);
    float*  esc    = (float*)take((size_t)E_EDGES * 4);
    int*    deg    = (int*)take(N_NODES * 4);            // counts, then cursor
    int*    rowst  = (int*)take((N_NODES + 1) * 4);
    int*    le     = (int*)take((size_t)E_EDGES * 4);
    int*    ls     = (int*)take((size_t)E_EDGES * 4);
    float*  gates  = (float*)take((size_t)N_GRAPHS * GATES * 4);
    float*  hstate = (float*)take((size_t)N_GRAPHS * HID * 4);
    float*  cstate = (float*)take((size_t)N_GRAPHS * HID * 4);
    __bf16* hbst   = (__bf16*)take((size_t)N_GRAPHS * HID * 2);

    // ---- 1) weight / input conversion to bf16 ----
    for (int l = 0; l < 4; ++l) {
        int kin = (l == 0) ? IN_DIM : HID;
        int n = HID * kin;
        cvt_f32_bf16<<<(n + 255) / 256, 256, 0, stream>>>(lyr_lin_w[l], wl_lin[l], n);
        cvt_f32_bf16<<<(n + 255) / 256, 256, 0, stream>>>(lyr_res_w[l], wl_res[l], n);
    }
    build_wcat<<<(GATES * KCAT) / 256, 256, 0, stream>>>(lstm_wih, lstm_whh, wcat);
    cvt_f32_bf16<<<(N_NODES * IN_DIM) / 256, 256, 0, stream>>>(x, actb, N_NODES * IN_DIM);

    // ---- 2) CSR of incoming edges (shared by all 4 layers) ----
    fill_i32<<<N_NODES / 256, 256, 0, stream>>>(deg, 0, N_NODES);
    csr_count<<<E_EDGES / 256, 256, 0, stream>>>(eidx, deg);
    csr_scan<<<1, 1024, 0, stream>>>(deg, rowst);
    fill_i32<<<N_NODES / 256, 256, 0, stream>>>(deg, 0, N_NODES);
    csr_scatter<<<E_EDGES / 256, 256, 0, stream>>>(eidx, rowst, deg, le, ls);

    // ---- 3) GAT layers ----
    int kin = IN_DIM;
    for (int l = 0; l < 4; ++l) {
        wmma_gemm_bt<<<dim3(N_NODES / 16, 1), 256, 0, stream>>>(
            actb, wl_lin[l], nullptr, hlin, N_NODES, HID, kin);
        wmma_gemm_bt<<<dim3(N_NODES / 16, 1), 256, 0, stream>>>(
            actb, wl_res[l], lyr_res_b[l], outres, N_NODES, HID, kin);
        gat_scores<<<N_NODES / 8, 256, 0, stream>>>(hlin, lyr_attn[l], s_i, s_j);
        gat_edge_scores<<<E_EDGES / 256, 256, 0, stream>>>(eidx, s_i, s_j, esc);
        gat_aggregate_ln_elu<<<N_NODES / 8, 256, 0, stream>>>(
            hlin, outres, esc, rowst, le, ls, lyr_g[l], lyr_b[l], actb);
        kin = HID;
    }

    // ---- 4) LSTM scan over SEQ steps (h0 = c0 = 0) ----
    fill_f32<<<(N_GRAPHS * HID) / 256, 256, 0, stream>>>(hstate, 0.f, N_GRAPHS * HID);
    fill_f32<<<(N_GRAPHS * HID) / 256, 256, 0, stream>>>(cstate, 0.f, N_GRAPHS * HID);
    fill_i32<<<(N_GRAPHS * HID / 2) / 256, 256, 0, stream>>>((int*)hbst, 0, N_GRAPHS * HID / 2);
    for (int t = 0; t < SEQ; ++t) {
        lstm_step_gemm<<<dim3(2, 4), 256, 0, stream>>>(
            actb, hbst, wcat, lstm_bih, lstm_bhh, gates, t);
        lstm_update<<<(N_GRAPHS * HID) / 256, 256, 0, stream>>>(gates, hstate, cstate, hbst);
    }

    // ---- 5) classifier ----
    classifier<<<NCLS, 1024, 0, stream>>>(hstate, fc_w, fc_b, out);
}
